// CompQGCNEncoder_38388417692355
// MI455X (gfx1250) — matrix operand order repaired and checked by reading, not compiled, not run on hardware
//
#include <hip/hip_runtime.h>
#include <hip/hip_bf16.h>

#define D   256      // feature dim
#define DP  264      // padded bf16 panel stride (528 B row => banks spread 4 mod 64)
#define KT  8        // K tiles of 32 (8*32 = 256)

typedef __attribute__((ext_vector_type(16))) __bf16 v16bf;
typedef __attribute__((ext_vector_type(8)))  __bf16 v8bf;
typedef __attribute__((ext_vector_type(8)))  float  v8f;

static constexpr float ALPHA  = 0.8f;
static constexpr float BN_EPS = 1e-5f;

// ---------------------------------------------------------------------------
// WMMA helpers.
// A layout (16-bit A 16x32): lane L holds row M=L%16; halves 0..7 -> K = kb+8*(L/16)+h,
// halves 8..15 -> K = kb+16+8*(L/16)+(h-8).  Panel is bf16 with stride DP, so each
// tile fetch is two 16-byte ds_load_b128, no conversions.
// B layout (16-bit B 32x16): lane L holds col N=L%16; half h -> same K as the A layout above,
// pre-packed contiguously per (kt,nt,lane).
// ---------------------------------------------------------------------------
__device__ __forceinline__ v16bf ldsA_panel(const __bf16* __restrict__ A, int lane, int kb) {
  const __bf16* p = A + (lane & 15) * DP + kb + (lane >> 4) * 8;
  v8bf lo = *(const v8bf*)p;          // K = kb + 8*hi  .. +7
  v8bf hi = *(const v8bf*)(p + 16);   // K = kb + 16 + 8*hi .. +7
  return __builtin_shufflevector(lo, hi, 0, 1, 2, 3, 4, 5, 6, 7,
                                         8, 9, 10, 11, 12, 13, 14, 15);
}

__device__ __forceinline__ v16bf loadB(const __bf16* __restrict__ wp, int kt, int nt, int lane) {
  return *(const v16bf*)(wp + ((size_t)((kt * 16 + nt) * 32 + lane) << 4));
}

// Wave-level 16x16 dual-tile GEMM: C[16,nt*16..] and C[16,(nt+8)*16..] over K=256.
// A tile loaded once per K-step, feeds two independent WMMAs.
__device__ __forceinline__ void wmma_gemm16x2(const __bf16* __restrict__ As,
                                              const __bf16* __restrict__ B,
                                              int lane, int wave, v8f& acc0, v8f& acc1) {
#pragma unroll
  for (int kt = 0; kt < KT; ++kt) {
    v16bf a  = ldsA_panel(As, lane, kt * 32);
    v16bf b0 = loadB(B, kt, wave, lane);
    v16bf b1 = loadB(B, kt, wave + 8, lane);
    acc0 = __builtin_amdgcn_wmma_f32_16x16x32_bf16(false, a, false, b0, (short)0, acc0, false, false);
    acc1 = __builtin_amdgcn_wmma_f32_16x16x32_bf16(false, a, false, b1, (short)0, acc1, false, false);
  }
}

// Pack a row-major [D x D] f32 weight (w[k*D+n]) into the bf16 B-matrix lane layout.
__global__ void pack_b_bf16(const float* __restrict__ w, __bf16* __restrict__ out) {
  int t = blockIdx.x * blockDim.x + threadIdx.x;   // 0 .. D*D-1
  int h    = t & 15;
  int lane = (t >> 4) & 31;
  int nt   = (t >> 9) & 15;
  int kt   = (t >> 13) & 7;
  int n = nt * 16 + (lane & 15);
  int k = kt * 32 + (h < 8 ? (lane >> 4) * 8 + h : 16 + (lane >> 4) * 8 + (h - 8));
  out[t] = (__bf16)w[(size_t)k * D + n];
}

// ---------------------------------------------------------------------------
// Degree / symmetric norm (two directed passes share edge storage)
// ---------------------------------------------------------------------------
__global__ void deg_kernel(const int* __restrict__ row, const int* __restrict__ col,
                           float* __restrict__ deg1, float* __restrict__ deg2, int E) {
  int t = blockIdx.x * blockDim.x + threadIdx.x;
  if (t < E) {
    atomicAdd(&deg1[row[t]], 1.0f);
    atomicAdd(&deg2[col[t]], 1.0f);
  }
}

__device__ __forceinline__ float dinv_f(float d) { return d > 0.0f ? rsqrtf(d) : 0.0f; }

__global__ void norm_kernel(const int* __restrict__ row, const int* __restrict__ col,
                            const float* __restrict__ deg1, const float* __restrict__ deg2,
                            float* __restrict__ norm1, float* __restrict__ norm2, int E) {
  int t = blockIdx.x * blockDim.x + threadIdx.x;
  if (t < E) {
    int s = row[t], d = col[t];
    norm1[t] = dinv_f(deg1[s]) * dinv_f(deg1[d]);   // in-pass: deg over src rows
    norm2[t] = dinv_f(deg2[d]) * dinv_f(deg2[s]);   // out-pass: deg over dst rows
  }
}

// ---------------------------------------------------------------------------
// Qualifier transform: q_emb = ccorr(x[q_ent], rel[q_rel]); scatter into coal[edge]
// One 256-thread block per qualifier; ccorr[k] = sum_j a[j] * b[(j+k)%D]
// ---------------------------------------------------------------------------
__global__ void qual_kernel(const float* __restrict__ x, const float* __restrict__ rel,
                            const int* __restrict__ qrel, const int* __restrict__ qent,
                            const int* __restrict__ qptr, float* __restrict__ coal,
                            int Q, int E) {
  __shared__ float a[D];
  __shared__ float b[2 * D];
  int q = blockIdx.x;            // 0 .. 2Q-1
  int t = threadIdx.x;           // 0 .. 255
  int ent = qent[q], r = qrel[q];
  a[t] = x[(size_t)ent * D + t];
  float bv = rel[(size_t)r * D + t];
  b[t] = bv; b[D + t] = bv;
  __syncthreads();
  float s = 0.0f;
#pragma unroll 8
  for (int j = 0; j < D; ++j) s = fmaf(a[j], b[j + t], s);
  size_t eg = (size_t)((q < Q) ? qptr[q] : E + qptr[q]);
  atomicAdd(&coal[eg * D + t], s);
}

// ---------------------------------------------------------------------------
// rel_hat[e] = ALPHA*rel[etype[e]] + (1-ALPHA) * (coal[e] @ w_q), written in place.
// 16-edge band per 256-thread block; bf16 A panel staged once.
// ---------------------------------------------------------------------------
__global__ void relhat_kernel(const float* __restrict__ rel, const int* __restrict__ etype,
                              const __bf16* __restrict__ wq, float* __restrict__ coal) {
  __shared__ __align__(16) __bf16 As[16][DP];
  int band = blockIdx.x, t = threadIdx.x;
  int e0 = band * 16;
  for (int i = 2 * t; i < 16 * D; i += 512) {
    int r = i >> 8, c = i & 255;
    float2 v = *(const float2*)&coal[(size_t)(e0 + r) * D + c];
    As[r][c] = (__bf16)v.x; As[r][c + 1] = (__bf16)v.y;
  }
  __syncthreads();
  int wave = t >> 5, lane = t & 31;
  v8f acc0 = {}, acc1 = {};
  wmma_gemm16x2(&As[0][0], wq, lane, wave, acc0, acc1);
#pragma unroll
  for (int half = 0; half < 2; ++half) {
    const v8f& acc = half ? acc1 : acc0;
    int n = (wave + half * 8) * 16 + (lane & 15);
#pragma unroll
    for (int r = 0; r < 8; ++r) {
      int m = r + 8 * (lane >> 4);
      int e = e0 + m;
      float base = rel[(size_t)etype[e] * D + n];
      coal[(size_t)e * D + n] = ALPHA * base + (1.0f - ALPHA) * acc[r];
    }
  }
}

// ---------------------------------------------------------------------------
// Fused edge kernel: h = ccorr(x[src], rel_hat)  ->  msg = (h @ w) * norm
// -> atomic scatter-add into node accumulator.  16 edges per block.
// ---------------------------------------------------------------------------
__global__ void edge_kernel(const float* __restrict__ x, const float* __restrict__ relhat,
                            const int* __restrict__ row, const int* __restrict__ col,
                            const float* __restrict__ norm1, const float* __restrict__ norm2,
                            const __bf16* __restrict__ win, const __bf16* __restrict__ wout,
                            float* __restrict__ accum, int E, int E2) {
  __shared__ float AX[16][D];                 // x[src] rows           (16 KB)
  __shared__ float BR[16][2 * D];             // rel_hat rows, doubled (32 KB)
  __shared__ __align__(16) __bf16 HH[16][DP]; // ccorr results, bf16   (8.25 KB)
  int band = blockIdx.x, t = threadIdx.x;
  int e0 = band * 16;
  if (e0 + 16 < E2) __builtin_prefetch(&relhat[(size_t)(e0 + 16) * D], 0, 1);
  for (int i = t; i < 16 * D; i += 256) {
    int r = i >> 8, c = i & 255;
    AX[r][c] = x[(size_t)row[e0 + r] * D + c];
    float bv = relhat[(size_t)(e0 + r) * D + c];
    BR[r][c] = bv; BR[r][D + c] = bv;
  }
  __syncthreads();
  {   // cross-correlation: thread -> (edge r, 16 consecutive outputs at k0)
    int r = t >> 4, k0 = (t & 15) * 16;
    float acc[16];
#pragma unroll
    for (int kk = 0; kk < 16; ++kk) acc[kk] = 0.0f;
    for (int j = 0; j < D; ++j) {
      float av = AX[r][j];
      const float* bp = &BR[r][j + k0];
#pragma unroll
      for (int kk = 0; kk < 16; ++kk) acc[kk] = fmaf(av, bp[kk], acc[kk]);
    }
#pragma unroll
    for (int kk = 0; kk < 16; ++kk) HH[r][k0 + kk] = (__bf16)acc[kk];
  }
  __syncthreads();
  const __bf16* w = (e0 < E) ? win : wout;   // E % 16 == 0: bands never straddle halves
  int wave = t >> 5, lane = t & 31;
  v8f acc0 = {}, acc1 = {};
  wmma_gemm16x2(&HH[0][0], w, lane, wave, acc0, acc1);
#pragma unroll
  for (int half = 0; half < 2; ++half) {
    const v8f& acc = half ? acc1 : acc0;
    int n = (wave + half * 8) * 16 + (lane & 15);
#pragma unroll
    for (int r = 0; r < 8; ++r) {
      int m = r + 8 * (lane >> 4);
      int e = e0 + m;
      float nm = (e < E) ? norm1[e] : norm2[e - E];
      atomicAdd(&accum[(size_t)col[e] * D + n], acc[r] * nm);
    }
  }
}

// ---------------------------------------------------------------------------
// Self-loop: CW[j,n] = sum_k loop_rel[(j+k)%D] * w_loop[k,n]  (shared circulant GEMM)
// ---------------------------------------------------------------------------
__global__ void cw_kernel(const float* __restrict__ loop_rel, const float* __restrict__ w_loop,
                          float* __restrict__ CW) {
  int t = blockIdx.x * blockDim.x + threadIdx.x;   // 0 .. D*D-1
  int j = t >> 8, n = t & 255;
  float s = 0.0f;
  for (int k = 0; k < D; ++k) s = fmaf(loop_rel[(j + k) & 255], w_loop[(size_t)k * D + n], s);
  CW[t] = s;
}

// loop_res = x @ CW, accumulated (plain add: runs after all atomics in stream order)
__global__ void loop_gemm_kernel(const float* __restrict__ x, const __bf16* __restrict__ cw,
                                 float* __restrict__ accum) {
  __shared__ __align__(16) __bf16 As[16][DP];
  int band = blockIdx.x, t = threadIdx.x;
  int e0 = band * 16;
  for (int i = 2 * t; i < 16 * D; i += 512) {
    int r = i >> 8, c = i & 255;
    float2 v = *(const float2*)&x[(size_t)(e0 + r) * D + c];
    As[r][c] = (__bf16)v.x; As[r][c + 1] = (__bf16)v.y;
  }
  __syncthreads();
  int wave = t >> 5, lane = t & 31;
  v8f acc0 = {}, acc1 = {};
  wmma_gemm16x2(&As[0][0], cw, lane, wave, acc0, acc1);
#pragma unroll
  for (int half = 0; half < 2; ++half) {
    const v8f& acc = half ? acc1 : acc0;
    int n = (wave + half * 8) * 16 + (lane & 15);
#pragma unroll
    for (int r = 0; r < 8; ++r) {
      int m = r + 8 * (lane >> 4);
      accum[(size_t)(e0 + m) * D + n] += acc[r];
    }
  }
}

// ---------------------------------------------------------------------------
// BatchNorm statistics over rows (columns = 256 = blockDim)
// ---------------------------------------------------------------------------
__global__ void stats_kernel(const float* __restrict__ accum, const float* __restrict__ bias,
                             float* __restrict__ sum, float* __restrict__ sumsq, int N) {
  int k = threadIdx.x;
  int r0 = blockIdx.x * 128;
  float s = 0.0f, s2 = 0.0f;
  for (int i = 0; i < 128; ++i) {
    int r = r0 + i;
    if (r < N) {
      float v = accum[(size_t)r * D + k] * (1.0f / 3.0f) + bias[k];
      s += v; s2 += v * v;
    }
  }
  atomicAdd(&sum[k], s);
  atomicAdd(&sumsq[k], s2);
}

__global__ void final_kernel(const float* __restrict__ accum, const float* __restrict__ bias,
                             const float* __restrict__ sum, const float* __restrict__ sumsq,
                             const float* __restrict__ gamma, const float* __restrict__ beta,
                             float* __restrict__ out, int N) {
  size_t t = (size_t)blockIdx.x * blockDim.x + threadIdx.x;
  if (t < (size_t)N * D) {
    int k = (int)(t & 255);
    float inv_n = 1.0f / (float)N;
    float mean = sum[k] * inv_n;
    float var  = sumsq[k] * inv_n - mean * mean;
    float v = accum[t] * (1.0f / 3.0f) + bias[k];
    v = (v - mean) * rsqrtf(var + BN_EPS) * gamma[k] + beta[k];
    out[t] = tanhf(v);
  }
}

// rel_out = rel_embed @ w_rel   (500 rows, guarded bands)
__global__ void rel_gemm_kernel(const float* __restrict__ rel, const __bf16* __restrict__ wr,
                                float* __restrict__ out, int R) {
  __shared__ __align__(16) __bf16 As[16][DP];
  int band = blockIdx.x, t = threadIdx.x;
  int e0 = band * 16;
  for (int i = 2 * t; i < 16 * D; i += 512) {
    int r = i >> 8, c = i & 255;
    if (e0 + r < R) {
      float2 v = *(const float2*)&rel[(size_t)(e0 + r) * D + c];
      As[r][c] = (__bf16)v.x; As[r][c + 1] = (__bf16)v.y;
    } else {
      As[r][c] = (__bf16)0.0f; As[r][c + 1] = (__bf16)0.0f;
    }
  }
  __syncthreads();
  int wave = t >> 5, lane = t & 31;
  v8f acc0 = {}, acc1 = {};
  wmma_gemm16x2(&As[0][0], wr, lane, wave, acc0, acc1);
#pragma unroll
  for (int half = 0; half < 2; ++half) {
    const v8f& acc = half ? acc1 : acc0;
    int n = (wave + half * 8) * 16 + (lane & 15);
#pragma unroll
    for (int r = 0; r < 8; ++r) {
      int m = r + 8 * (lane >> 4);
      if (e0 + m < R) out[(size_t)(e0 + m) * D + n] = acc[r];
    }
  }
}

// ---------------------------------------------------------------------------
extern "C" void kernel_launch(void* const* d_in, const int* in_sizes, int n_in,
                              void* d_out, int out_size, void* d_ws, size_t ws_size,
                              hipStream_t stream) {
  const float* x        = (const float*)d_in[0];
  const float* rel      = (const float*)d_in[1];
  const float* w_in     = (const float*)d_in[2];
  const float* w_out    = (const float*)d_in[3];
  const float* w_loop   = (const float*)d_in[4];
  const float* w_rel    = (const float*)d_in[5];
  const float* w_q      = (const float*)d_in[6];
  const float* loop_rel = (const float*)d_in[7];
  const float* bias     = (const float*)d_in[8];
  const float* bn_gamma = (const float*)d_in[9];
  const float* bn_beta  = (const float*)d_in[10];
  const int*   ei       = (const int*)d_in[11];   // [2, 2E] row-major
  const int*   etype    = (const int*)d_in[12];   // [2E]
  const int*   quals    = (const int*)d_in[13];   // [3, 2Q] row-major

  const int N  = in_sizes[0] / D;
  const int R  = in_sizes[1] / D;
  const int E2 = in_sizes[12];
  const int E  = E2 / 2;
  const int Q2 = in_sizes[13] / 3;
  const int Q  = Q2 / 2;

  const int* row  = ei;            // edge_index[0,:]
  const int* colp = ei + E2;       // edge_index[1,:]
  const int* qrel = quals;
  const int* qent = quals + Q2;
  const int* qptr = quals + 2 * Q2;

  // ---- workspace carve-out ----
  size_t off = 0;
  auto carve = [&](size_t bytes) -> void* {
    void* p = (char*)d_ws + off;
    off = (off + bytes + 255) & ~(size_t)255;
    return p;
  };
  float*  coal   = (float*)carve((size_t)E2 * D * sizeof(float));   // coalesced -> rel_hat (in place)
  float*  accum  = (float*)carve((size_t)N * D * sizeof(float));
  float*  deg1   = (float*)carve((size_t)N * sizeof(float));
  float*  deg2   = (float*)carve((size_t)N * sizeof(float));
  float*  norm1  = (float*)carve((size_t)E * sizeof(float));
  float*  norm2  = (float*)carve((size_t)E * sizeof(float));
  float*  CW     = (float*)carve((size_t)D * D * sizeof(float));
  float*  colsum = (float*)carve((size_t)D * sizeof(float));
  float*  colsq  = (float*)carve((size_t)D * sizeof(float));
  __bf16* p_wq   = (__bf16*)carve((size_t)D * D * sizeof(__bf16));
  __bf16* p_win  = (__bf16*)carve((size_t)D * D * sizeof(__bf16));
  __bf16* p_wout = (__bf16*)carve((size_t)D * D * sizeof(__bf16));
  __bf16* p_cw   = (__bf16*)carve((size_t)D * D * sizeof(__bf16));
  __bf16* p_wrel = (__bf16*)carve((size_t)D * D * sizeof(__bf16));
  (void)ws_size; (void)n_in; (void)out_size;

  float* ent_out = (float*)d_out;
  float* rel_out = ent_out + (size_t)N * D;

  // ---- zero accumulators ----
  hipMemsetAsync(coal,   0, (size_t)E2 * D * sizeof(float), stream);
  hipMemsetAsync(accum,  0, (size_t)N * D * sizeof(float), stream);
  hipMemsetAsync(deg1,   0, (size_t)N * sizeof(float), stream);
  hipMemsetAsync(deg2,   0, (size_t)N * sizeof(float), stream);
  hipMemsetAsync(colsum, 0, (size_t)D * sizeof(float), stream);
  hipMemsetAsync(colsq,  0, (size_t)D * sizeof(float), stream);

  // ---- pack weights to bf16 WMMA-B layout ----
  pack_b_bf16<<<D * D / 256, 256, 0, stream>>>(w_q,   p_wq);
  pack_b_bf16<<<D * D / 256, 256, 0, stream>>>(w_in,  p_win);
  pack_b_bf16<<<D * D / 256, 256, 0, stream>>>(w_out, p_wout);
  pack_b_bf16<<<D * D / 256, 256, 0, stream>>>(w_rel, p_wrel);

  // ---- degree / norm ----
  deg_kernel<<<(E + 255) / 256, 256, 0, stream>>>(row, colp, deg1, deg2, E);
  norm_kernel<<<(E + 255) / 256, 256, 0, stream>>>(row, colp, deg1, deg2, norm1, norm2, E);

  // ---- qualifier scatter ----
  qual_kernel<<<Q2, 256, 0, stream>>>(x, rel, qrel, qent, qptr, coal, Q, E);

  // ---- rel_hat (in place over coal) ----
  relhat_kernel<<<E2 / 16, 256, 0, stream>>>(rel, etype, p_wq, coal);

  // ---- fused per-edge ccorr + GEMM + scatter ----
  edge_kernel<<<E2 / 16, 256, 0, stream>>>(x, coal, row, colp, norm1, norm2,
                                           p_win, p_wout, accum, E, E2);

  // ---- self loop ----
  cw_kernel<<<D * D / 256, 256, 0, stream>>>(loop_rel, w_loop, CW);
  pack_b_bf16<<<D * D / 256, 256, 0, stream>>>(CW, p_cw);
  loop_gemm_kernel<<<N / 16, 256, 0, stream>>>(x, p_cw, accum);

  // ---- batch norm + tanh ----
  stats_kernel<<<(N + 127) / 128, 256, 0, stream>>>(accum, bias, colsum, colsq, N);
  final_kernel<<<(int)(((size_t)N * D + 255) / 256), 256, 0, stream>>>(
      accum, bias, colsum, colsq, bn_gamma, bn_beta, ent_out, N);

  // ---- relation output ----
  rel_gemm_kernel<<<(R + 15) / 16, 256, 0, stream>>>(rel, p_wrel, rel_out, R);
}
